// MT_GAT_topk_shareEn_multiple8_joint_cross_70712341561398
// MI455X (gfx1250) — matrix operand order repaired and checked by reading, not compiled, not run on hardware
//
#include <hip/hip_runtime.h>
#include <hip/hip_bf16.h>
#include <math.h>

// ---------------------------------------------------------------------------
// Problem constants (from reference)
// ---------------------------------------------------------------------------
#define NGRAPH 128
#define NPG    264
#define NTOT   (NGRAPH*NPG)        // 33792 nodes
#define DEG    16
#define NEDGE  (NTOT*DEG)          // 540672 edges
#define NLOOP  (NEDGE+NTOT)        // + self loops (GAT)
#define HDIM   64
#define DDIM   264
#define HCDIM  128
#define KPOOL  132

typedef __attribute__((ext_vector_type(16))) _Float16 v16h;
typedef __attribute__((ext_vector_type(8)))  float    v8f;

// LDS byte offset (relative to wave LDS base) of a __shared__ object:
// generic -> addrspace(3) cast keeps the low 32 bits, which is the LDS offset.
typedef __attribute__((address_space(3))) char* lds_cptr;
__device__ __forceinline__ unsigned lds_off(const void* p) {
  return (unsigned)(size_t)(lds_cptr)(p);
}
__device__ __forceinline__ void async_b128(const void* gp, unsigned loff) {
  asm volatile("global_load_async_to_lds_b128 %0, %1, off"
               :: "v"(loff), "v"(gp) : "memory");
}
__device__ __forceinline__ void wait_async0() {
  asm volatile("s_wait_asynccnt 0x0" ::: "memory");
}

// ---------------------------------------------------------------------------
// Generic batched WMMA GEMM:  C = epilogue( alpha * (A [+Aadd]) @ op(B) )
//   op(B) = B^T when bT (B stored [N,K] row-major, i.e. torch Linear weight)
//   epilogue: +bias[n], +add_pre[m,n], BN-eval(g,b), ReLU, accumulate-into-C
// 256 threads = 8 waves (4m x 2n); block tile 128x64; per wave 32x32 patch
// = 2x2 fragments = 4 v_wmma per k-tile. Tiles staged to LDS with
// global_load_async_to_lds_b128 when fully in-bounds; edge tiles use
// branch-free clamped loads with value select (zero fill).
// ---------------------------------------------------------------------------
#define TM 128
#define TN 64
#define TK 32
#define ASTR 36   // LDS row stride in floats: 144B, keeps 16B chunks aligned

__global__ void __launch_bounds__(256)
wmma_gemm_f32(const float* __restrict__ A, const float* __restrict__ Aadd,
              long lda, long sA,
              const float* __restrict__ Bm, long ldb, long sB, int bT,
              float* __restrict__ C, long ldc, long sC,
              int M, int Nn, int Kd, float alpha,
              const float* __restrict__ bias,
              const float* __restrict__ addp, long ldadd,
              const float* __restrict__ bng, const float* __restrict__ bnb,
              int relu, int accum)
{
  __shared__ float As[TM * ASTR];   // 18432 B
  __shared__ float Bs[TN * ASTR];   //  9216 B

  const long z = blockIdx.z;
  A  += z * sA;
  Bm += z * sB;
  C  += z * sC;
  const float* Aa = Aadd ? (Aadd + z * sA) : nullptr;
  const float* Ap = addp ? (addp + z * sC) : nullptr;  // residual batched like C

  const int tid  = threadIdx.x;
  const int w    = tid >> 5;
  const int lane = tid & 31;
  const int g    = lane >> 4;
  const int lm   = lane & 15;
  const int mBase = blockIdx.x * TM;
  const int nBase = blockIdx.y * TN;
  const int wmRow = (w >> 1) * 32;   // wave's 32-row slot in the 128-row tile
  const int wnCol = (w & 1) * 32;    // wave's 32-col slot in the 64-col tile

  v8f acc00 = {}, acc01 = {}, acc10 = {}, acc11 = {};

  for (int k0 = 0; k0 < Kd; k0 += TK) {
    // ---------- stage A tile [128 x 32]: 1024 16B-chunks, 4 per thread ------
    const bool fullA = (mBase + TM <= M) && (k0 + TK <= Kd) && (Aa == nullptr);
#pragma unroll
    for (int p = 0; p < 4; ++p) {
      const int q   = tid + 256 * p;      // 0..1023
      const int row = q >> 3;             // 0..127
      const int c4  = (q & 7) * 4;        // 0,4,...,28
      if (fullA) {
        const float* gp = A + (long)(mBase + row) * lda + (k0 + c4);
        async_b128(gp, lds_off(&As[row * ASTR + c4]));
      } else {
        const int m  = mBase + row;
        const int mc = m < M ? m : (M - 1);
#pragma unroll
        for (int j = 0; j < 4; ++j) {
          const int k  = k0 + c4 + j;
          const int kc = k < Kd ? k : (Kd - 1);
          float v = A[(long)mc * lda + kc];
          if (Aa) v += Aa[(long)mc * lda + kc];
          As[row * ASTR + c4 + j] = (m < M && k < Kd) ? v : 0.f;
        }
      }
    }
    // ---------- stage B tile [64 x 32] (rows=n if bT, rows=k if !bT) --------
    const bool fullB = (k0 + TK <= Kd) && (nBase + TN <= Nn);
#pragma unroll
    for (int p = 0; p < 2; ++p) {
      const int q   = tid + 256 * p;      // 0..511
      const int row = q >> 3;             // 0..63
      const int c4  = (q & 7) * 4;
      if (fullB) {
        const float* gp = bT ? Bm + (long)(nBase + row) * ldb + (k0 + c4)
                             : Bm + (long)(k0 + row) * ldb + (nBase + c4);
        async_b128(gp, lds_off(&Bs[row * ASTR + c4]));
      } else {
#pragma unroll
        for (int j = 0; j < 4; ++j) {
          float v; bool ok;
          if (bT) {
            const int n = nBase + row, k = k0 + c4 + j;
            const int nc = n < Nn ? n : (Nn - 1), kc = k < Kd ? k : (Kd - 1);
            v = Bm[(long)nc * ldb + kc];
            ok = (n < Nn && k < Kd);
          } else {
            const int k = k0 + row, n = nBase + c4 + j;
            const int kc = k < Kd ? k : (Kd - 1), nc = n < Nn ? n : (Nn - 1);
            v = Bm[(long)kc * ldb + nc];
            ok = (k < Kd && n < Nn);
          }
          Bs[row * ASTR + c4 + j] = ok ? v : 0.f;
        }
      }
    }
    wait_async0();
    __syncthreads();

    // ---------- fragments from LDS (float4 reads; 16B-aligned) --------------
    v16h afr[2], bfr[2];
#pragma unroll
    for (int i = 0; i < 2; ++i) {
      const float* base = &As[(wmRow + 16 * i + lm) * ASTR];
      const float4 u0 = *(const float4*)(base + 16 * g);
      const float4 u1 = *(const float4*)(base + 16 * g + 4);
      const float4 u2 = *(const float4*)(base + 16 * g + 8);
      const float4 u3 = *(const float4*)(base + 16 * g + 12);
      afr[i][0]=(_Float16)u0.x; afr[i][1]=(_Float16)u0.y; afr[i][2]=(_Float16)u0.z; afr[i][3]=(_Float16)u0.w;
      afr[i][4]=(_Float16)u1.x; afr[i][5]=(_Float16)u1.y; afr[i][6]=(_Float16)u1.z; afr[i][7]=(_Float16)u1.w;
      afr[i][8]=(_Float16)u2.x; afr[i][9]=(_Float16)u2.y; afr[i][10]=(_Float16)u2.z; afr[i][11]=(_Float16)u2.w;
      afr[i][12]=(_Float16)u3.x; afr[i][13]=(_Float16)u3.y; afr[i][14]=(_Float16)u3.z; afr[i][15]=(_Float16)u3.w;
    }
#pragma unroll
    for (int jf = 0; jf < 2; ++jf) {
      const int colB = wnCol + 16 * jf + lm;   // 0..63
      if (bT) {
        const float* base = &Bs[colB * ASTR + 16 * g];
        const float4 u0 = *(const float4*)(base);
        const float4 u1 = *(const float4*)(base + 4);
        const float4 u2 = *(const float4*)(base + 8);
        const float4 u3 = *(const float4*)(base + 12);
        bfr[jf][0]=(_Float16)u0.x; bfr[jf][1]=(_Float16)u0.y; bfr[jf][2]=(_Float16)u0.z; bfr[jf][3]=(_Float16)u0.w;
        bfr[jf][4]=(_Float16)u1.x; bfr[jf][5]=(_Float16)u1.y; bfr[jf][6]=(_Float16)u1.z; bfr[jf][7]=(_Float16)u1.w;
        bfr[jf][8]=(_Float16)u2.x; bfr[jf][9]=(_Float16)u2.y; bfr[jf][10]=(_Float16)u2.z; bfr[jf][11]=(_Float16)u2.w;
        bfr[jf][12]=(_Float16)u3.x; bfr[jf][13]=(_Float16)u3.y; bfr[jf][14]=(_Float16)u3.z; bfr[jf][15]=(_Float16)u3.w;
      } else {
#pragma unroll
        for (int e = 0; e < 16; ++e)
          bfr[jf][e] = (_Float16)Bs[(e + 16 * g) * ASTR + colB];
      }
    }
    acc00 = __builtin_amdgcn_wmma_f32_16x16x32_f16(false, afr[0], false, bfr[0], (short)0, acc00, false, false);
    acc01 = __builtin_amdgcn_wmma_f32_16x16x32_f16(false, afr[0], false, bfr[1], (short)0, acc01, false, false);
    acc10 = __builtin_amdgcn_wmma_f32_16x16x32_f16(false, afr[1], false, bfr[0], (short)0, acc10, false, false);
    acc11 = __builtin_amdgcn_wmma_f32_16x16x32_f16(false, afr[1], false, bfr[1], (short)0, acc11, false, false);
    __syncthreads();
  }

  // ---------- epilogue ----------
  const v8f* accs[4] = {&acc00, &acc01, &acc10, &acc11};
#pragma unroll
  for (int fi = 0; fi < 2; ++fi) {
#pragma unroll
    for (int fj = 0; fj < 2; ++fj) {
      const v8f a = *accs[fi * 2 + fj];
#pragma unroll
      for (int i = 0; i < 8; ++i) {
        const int m = mBase + wmRow + 16 * fi + i + 8 * g;  // C layout: VGPR i, half g
        const int n = nBase + wnCol + 16 * fj + lm;
        if (m < M && n < Nn) {
          float v = a[i] * alpha;
          if (bias) v += bias[n];
          if (Ap)   v += Ap[(long)m * ldadd + n];
          if (bng)  v = v * (bng[n] * 0.9999950000374997f) + bnb[n]; // BN eval
          if (relu) v = fmaxf(v, 0.f);
          const long idx = (long)m * ldc + n;
          C[idx] = accum ? (C[idx] + v) : v;
        }
      }
    }
  }
}

// ---------------------------------------------------------------------------
// Fills
// ---------------------------------------------------------------------------
__global__ void fill_f(float* p, float v, long n) {
  long i = (long)blockIdx.x * blockDim.x + threadIdx.x;
  if (i < n) p[i] = v;
}
__global__ void fill_u(unsigned* p, unsigned v, long n) {
  long i = (long)blockIdx.x * blockDim.x + threadIdx.x;
  if (i < n) p[i] = v;
}

// ---------------------------------------------------------------------------
// GINE: aggr[dst] += relu(x[src] + (ew*We + be)), 8 channels per thread
// ---------------------------------------------------------------------------
__global__ void gine_msg(const float* __restrict__ x, const int* __restrict__ ei,
                         const float* __restrict__ ew, const float* __restrict__ We,
                         const float* __restrict__ be, float* __restrict__ aggr) {
  long tid = (long)blockIdx.x * blockDim.x + threadIdx.x;
  if (tid >= (long)NEDGE * 33) return;
  int e  = (int)(tid / 33);
  int c0 = (int)(tid % 33) * 8;
  int s = ei[e], d = ei[NEDGE + e];
  float w = ew[e];
#pragma unroll
  for (int j = 0; j < 8; ++j) {
    int c = c0 + j;
    float m = x[(long)s * DDIM + c] + (w * We[c] + be[c]);
    atomicAdd(&aggr[(long)d * DDIM + c], fmaxf(m, 0.f));
  }
}

// ---------------------------------------------------------------------------
// Row softmax over 264 cols; optional averaged-attention accumulation
// ---------------------------------------------------------------------------
__global__ void __launch_bounds__(256)
softmax264(float* __restrict__ S, float* __restrict__ att, float attw) {
  const long row = blockIdx.x;
  float* p = S + row * (long)NPG;
  const int t = threadIdx.x;
  __shared__ float red[256];
  float mv = -1e30f;
  for (int c = t; c < NPG; c += 256) mv = fmaxf(mv, p[c]);
  red[t] = mv; __syncthreads();
  for (int s = 128; s > 0; s >>= 1) { if (t < s) red[t] = fmaxf(red[t], red[t + s]); __syncthreads(); }
  mv = red[0]; __syncthreads();
  float sum = 0.f;
  for (int c = t; c < NPG; c += 256) { float e = __expf(p[c] - mv); p[c] = e; sum += e; }
  red[t] = sum; __syncthreads();
  for (int s = 128; s > 0; s >>= 1) { if (t < s) red[t] += red[t + s]; __syncthreads(); }
  const float inv = 1.f / red[0];
  for (int c = t; c < NPG; c += 256) {
    float v = p[c] * inv;
    p[c] = v;
    if (att) att[row * (long)NPG + c] += v * attw;  // each (row,c) owned by one thread
  }
}

// ---------------------------------------------------------------------------
// GAT helpers (segment softmax over edges via encoded-float atomic max)
// ---------------------------------------------------------------------------
__device__ __forceinline__ unsigned fenc(float f) {
  int i = __float_as_int(f);
  return (unsigned)(i >= 0 ? (i | 0x80000000) : ~i);
}
__device__ __forceinline__ float fdec(unsigned u) {
  int i = (u & 0x80000000u) ? (int)(u & 0x7fffffffu) : ~(int)u;
  return __int_as_float(i);
}

__global__ void gat_nodescore(const float* __restrict__ xp, const float* __restrict__ asrc,
                              const float* __restrict__ adst,
                              float* __restrict__ as_, float* __restrict__ ad_) {
  int n = blockIdx.x * blockDim.x + threadIdx.x;
  if (n >= NTOT) return;
  float s = 0.f, d = 0.f;
  for (int c = 0; c < HDIM; ++c) { float v = xp[(long)n * HDIM + c]; s += v * asrc[c]; d += v * adst[c]; }
  as_[n] = s; ad_[n] = d;
}

__global__ void gat_logit(const int* __restrict__ ei, const float* __restrict__ as_,
                          const float* __restrict__ ad_, float* __restrict__ logit,
                          unsigned* __restrict__ mEnc) {
  int el = blockIdx.x * blockDim.x + threadIdx.x;
  if (el >= NLOOP) return;
  int s, d;
  if (el < NEDGE) { s = ei[el]; d = ei[NEDGE + el]; } else { s = d = el - NEDGE; }
  float l = as_[s] + ad_[d];
  l = (l > 0.f) ? l : 0.2f * l;          // leaky_relu(0.2)
  logit[el] = l;
  atomicMax(&mEnc[d], fenc(l));
}

__global__ void gat_exp(const int* __restrict__ ei, float* __restrict__ exbuf,
                        const unsigned* __restrict__ mEnc, float* __restrict__ denom) {
  int el = blockIdx.x * blockDim.x + threadIdx.x;
  if (el >= NLOOP) return;
  int d = (el < NEDGE) ? ei[NEDGE + el] : (el - NEDGE);
  float ex = __expf(exbuf[el] - fdec(mEnc[d]));
  exbuf[el] = ex;
  atomicAdd(&denom[d], ex);
}

__global__ void gat_accum(const int* __restrict__ ei, const float* __restrict__ exbuf,
                          const float* __restrict__ denom, const float* __restrict__ xp,
                          float* __restrict__ acc) {
  long tid = (long)blockIdx.x * blockDim.x + threadIdx.x;
  if (tid >= (long)NLOOP * 8) return;
  int el = (int)(tid >> 3);
  int c0 = (int)(tid & 7) * 8;
  int s, d;
  if (el < NEDGE) { s = ei[el]; d = ei[NEDGE + el]; } else { s = d = el - NEDGE; }
  float a = exbuf[el] / (denom[d] + 1e-16f);
#pragma unroll
  for (int j = 0; j < 8; ++j)
    atomicAdd(&acc[(long)d * HDIM + c0 + j], a * xp[(long)s * HDIM + c0 + j]);
}

__global__ void gat_finish(float* __restrict__ acc, const float* __restrict__ bias) {
  long i = (long)blockIdx.x * blockDim.x + threadIdx.x;
  if (i >= (long)NTOT * HDIM) return;
  acc[i] = fmaxf(acc[i] + bias[i % HDIM], 0.f);
}

// ---------------------------------------------------------------------------
// Elementwise add
// ---------------------------------------------------------------------------
__global__ void addvec(const float* a, const float* b, float* o, long n) {
  long i = (long)blockIdx.x * blockDim.x + threadIdx.x;
  if (i < n) o[i] = a[i] + b[i];
}

// ---------------------------------------------------------------------------
// TopK pooling
// ---------------------------------------------------------------------------
__global__ void pool_score(const float* __restrict__ in, const float* __restrict__ w,
                           float* __restrict__ sc) {
  int n = blockIdx.x * blockDim.x + threadIdx.x;
  if (n >= NTOT) return;
  float nw = 0.f, d = 0.f;
  for (int c = 0; c < HDIM; ++c) { nw += w[c] * w[c]; d += in[(long)n * HDIM + c] * w[c]; }
  sc[n] = 1.f / (1.f + __expf(-d * rsqrtf(nw)));
}

// Stable descending rank (= torch.topk / lax.top_k tie-breaking).
__global__ void __launch_bounds__(288)
topk_graph(const float* __restrict__ sc, int* __restrict__ idx, float* __restrict__ vals,
           float* __restrict__ perm_out, float* __restrict__ sco_out) {
  int b = blockIdx.x, t = threadIdx.x;
  __shared__ float s[NPG];
  if (t < NPG) s[t] = sc[(long)b * NPG + t];
  __syncthreads();
  if (t < NPG) {
    float v = s[t];
    int r = 0;
    for (int j = 0; j < NPG; ++j) {
      float u = s[j];
      r += (u > v) || (u == v && j < t);
    }
    if (r < KPOOL) {
      idx[b * KPOOL + r]      = t;
      vals[b * KPOOL + r]     = v;
      perm_out[b * KPOOL + r] = (float)(b * NPG + t);
      sco_out[b * KPOOL + r]  = 1.f / (1.f + __expf(-v));   // double sigmoid
    }
  }
}

__global__ void gather_flat(const float* __restrict__ in, const int* __restrict__ idx,
                            const float* __restrict__ vals, float* __restrict__ flat) {
  long tid = (long)blockIdx.x * blockDim.x + threadIdx.x;
  if (tid >= (long)NGRAPH * KPOOL * HDIM) return;
  int c = (int)(tid & 63);
  long br = tid >> 6;
  int r = (int)(br % KPOOL), b = (int)(br / KPOOL);
  int i = idx[b * KPOOL + r];
  float v = in[((long)b * NPG + i) * HDIM + c] * vals[b * KPOOL + r];
  flat[((long)b * KPOOL + r) * HDIM + c] = fmaxf(v, 0.f);   // relu(flat) pre-applied
}

__global__ void lin_final1(const float* __restrict__ yh, const float* __restrict__ w,
                           const float* __restrict__ bias, float* __restrict__ out) {
  int b = threadIdx.x;
  if (b >= NGRAPH) return;
  float v = bias[0];
  for (int c = 0; c < HCDIM; ++c) v += yh[(long)b * HCDIM + c] * w[c];
  out[b] = v;
}

__global__ void lin_final2(const float* __restrict__ yh, const float* __restrict__ w,
                           const float* __restrict__ bias, float* __restrict__ out) {
  int b = threadIdx.x;
  if (b >= NGRAPH) return;
  float v0 = bias[0], v1 = bias[1];
  for (int c = 0; c < HCDIM; ++c) {
    float h = yh[(long)b * HCDIM + c];
    v0 += h * w[c];
    v1 += h * w[HCDIM + c];
  }
  float mx = fmaxf(v0, v1);
  float e0 = __expf(v0 - mx), e1 = __expf(v1 - mx);
  float s = e0 + e1;
  out[b * 2]     = e0 / s;
  out[b * 2 + 1] = e1 / s;
}

// ---------------------------------------------------------------------------
// Host orchestration
// ---------------------------------------------------------------------------
static inline void gemm(hipStream_t st,
                        const float* A, const float* Aadd, long lda, long sA,
                        const float* Bm, long ldb, long sB, int bT,
                        float* C, long ldc, long sC, int batch,
                        int M, int Nn, int Kd, float alpha,
                        const float* bias, const float* addp, long ldadd,
                        const float* bng, const float* bnb, int relu, int accum) {
  dim3 g((M + TM - 1) / TM, (Nn + TN - 1) / TN, batch);
  wmma_gemm_f32<<<g, 256, 0, st>>>(A, Aadd, lda, sA, Bm, ldb, sB, bT,
                                   C, ldc, sC, M, Nn, Kd, alpha,
                                   bias, addp, ldadd, bng, bnb, relu, accum);
}
static inline dim3 g1d(long n, int bs = 256) { return dim3((unsigned)((n + bs - 1) / bs)); }

extern "C" void kernel_launch(void* const* d_in, const int* in_sizes, int n_in,
                              void* d_out, int out_size, void* d_ws, size_t ws_size,
                              hipStream_t stream) {
  // ---- inputs (setup_inputs dict order, params flattened in insertion order)
  const float* x       = (const float*)d_in[0];
  const int*   ei      = (const int*)  d_in[1];
  const float* ew      = (const float*)d_in[2];
  const float* gine_ew = (const float*)d_in[4];
  const float* gine_eb = (const float*)d_in[5];
  const float* gine_w1 = (const float*)d_in[6];
  const float* gine_b1 = (const float*)d_in[7];
  const float* gine_w2 = (const float*)d_in[8];
  const float* gine_b2 = (const float*)d_in[9];
  const float* gps_iw  = (const float*)d_in[10];
  const float* gps_ib  = (const float*)d_in[11];
  const float* gps_ow  = (const float*)d_in[12];
  const float* gps_ob  = (const float*)d_in[13];
  const float* mlp_w1  = (const float*)d_in[14];
  const float* mlp_b1  = (const float*)d_in[15];
  const float* mlp_w2  = (const float*)d_in[16];
  const float* mlp_b2  = (const float*)d_in[17];
  const float* bn1g = (const float*)d_in[18], *bn1b = (const float*)d_in[19];
  const float* bn2g = (const float*)d_in[20], *bn2b = (const float*)d_in[21];
  const float* bn3g = (const float*)d_in[22], *bn3b = (const float*)d_in[23];
  const float* gatw[2]  = {(const float*)d_in[24], (const float*)d_in[28]};
  const float* gatas[2] = {(const float*)d_in[25], (const float*)d_in[29]};
  const float* gatad[2] = {(const float*)d_in[26], (const float*)d_in[30]};
  const float* gatb[2]  = {(const float*)d_in[27], (const float*)d_in[31]};
  const float* mha_iw = (const float*)d_in[32];
  const float* mha_ib = (const float*)d_in[33];
  const float* mha_ow = (const float*)d_in[34];
  const float* mha_ob = (const float*)d_in[35];
  const float* pool_w[2] = {(const float*)d_in[36], (const float*)d_in[37]};
  const float* lin_w[2]  = {(const float*)d_in[38], (const float*)d_in[42]};
  const float* lin_b[2]  = {(const float*)d_in[39], (const float*)d_in[43]};
  const float* fin_w[2]  = {(const float*)d_in[40], (const float*)d_in[44]};
  const float* fin_b[2]  = {(const float*)d_in[41], (const float*)d_in[45]};

  // ---- output layout (tuple concatenated flat, all as f32)
  float* out = (float*)d_out;
  const long o_y1 = 0, o_p1 = 128, o_s1 = 17024, o_y2 = 33920,
             o_p2 = 34176, o_s2 = 51072, o_att = 67968;

  // ---- workspace layout (floats)
  const long ND = (long)NTOT * DDIM;        // 8,921,088
  const long NH = (long)NTOT * HDIM;        // 2,162,688
  float* f = (float*)d_ws;
  float* aggr = f;                 // [ND]
  float* t1   = f + ND;            // [ND]  (also attn concat)
  float* hloc = f + 2 * ND;        // [ND]  (also out, x0)
  float* qkv  = f + 3 * ND;        // [3*ND]
  float* big  = f + 6 * ND;        // [2*ND] (mlp hidden; also per-head scores)
  float* s0   = f + 8 * ND;
  float* xp1  = s0;                float* xp2 = xp1 + NH;
  float* x11  = xp2 + NH;          float* x22 = x11 + NH;
  float* q2   = x22 + NH;          float* all1 = q2 + NH;
  float* inb  = all1 + NH;         // [NH] all1pre / pooled inputs
  float* kv2  = inb + NH;          // [2*NH]
  float* as_  = kv2 + 2 * NH;      float* ad_ = as_ + NTOT;
  float* denom = ad_ + NTOT;
  unsigned* mEnc = (unsigned*)(denom + NTOT);
  float* exbuf = denom + 2 * NTOT;                 // [NLOOP]
  float* vals  = exbuf + NLOOP;                    // [B*K]
  int*   idxb  = (int*)(vals + NGRAPH * KPOOL);    // [B*K]
  float* flat  = vals + 2 * (long)NGRAPH * KPOOL;  // [B*K*64]
  float* yh    = flat + (long)NGRAPH * KPOOL * HDIM; // [B*128]
  (void)ws_size; (void)n_in; (void)in_sizes; (void)out_size;

  const long sG = (long)NPG * DDIM;    // per-graph stride in [N,264]
  const long sS = (long)NPG * NPG;     // per-graph score stride

  // ============ GINEConv local branch ============
  fill_f<<<g1d(ND), 256, 0, stream>>>(aggr, 0.f, ND);
  gine_msg<<<g1d((long)NEDGE * 33), 256, 0, stream>>>(x, ei, ew, gine_ew, gine_eb, aggr);
  // t1 = relu((x+aggr) @ W1^T + b1)
  gemm(stream, x, aggr, DDIM, 0, gine_w1, DDIM, 0, 1, t1, DDIM, 0, 1,
       NTOT, DDIM, DDIM, 1.f, gine_b1, nullptr, 0, nullptr, nullptr, 1, 0);
  // hloc = BN1(t1 @ W2^T + b2 + x)
  gemm(stream, t1, nullptr, DDIM, 0, gine_w2, DDIM, 0, 1, hloc, DDIM, 0, 1,
       NTOT, DDIM, DDIM, 1.f, gine_b2, x, DDIM, bn1g, bn1b, 0, 0);

  // ============ GPS global attention (2 heads, hd=132) ============
  gemm(stream, x, nullptr, DDIM, 0, gps_iw, DDIM, 0, 1, qkv, 3 * DDIM, 0, 1,
       NTOT, 3 * DDIM, DDIM, 1.f, gps_ib, nullptr, 0, nullptr, nullptr, 0, 0);
  for (int h = 0; h < 2; ++h) {
    const float* Q  = qkv + h * 132;
    const float* Kp = qkv + DDIM + h * 132;
    const float* V  = qkv + 2 * DDIM + h * 132;
    gemm(stream, Q, nullptr, 3 * DDIM, (long)NPG * 3 * DDIM,
         Kp, 3 * DDIM, (long)NPG * 3 * DDIM, 1,
         big, NPG, sS, NGRAPH, NPG, NPG, 132, 0.08703882797784892f,
         nullptr, nullptr, 0, nullptr, nullptr, 0, 0);
    softmax264<<<NTOT, 256, 0, stream>>>(big, nullptr, 0.f);
    gemm(stream, big, nullptr, NPG, sS,
         V, 3 * DDIM, (long)NPG * 3 * DDIM, 0,
         t1 + h * 132, DDIM, sG, NGRAPH, NPG, 132, NPG, 1.f,
         nullptr, nullptr, 0, nullptr, nullptr, 0, 0);
  }
  // hloc += BN2(attn @ Wout^T + bout + x)   -> hloc = h + h2 = "out"
  gemm(stream, t1, nullptr, DDIM, 0, gps_ow, DDIM, 0, 1, hloc, DDIM, 0, 1,
       NTOT, DDIM, DDIM, 1.f, gps_ob, x, DDIM, bn2g, bn2b, 0, 1);
  // MLP: big = relu(out @ w1^T + b1); hloc = relu(BN3(out + big @ w2^T + b2)) = x0
  gemm(stream, hloc, nullptr, DDIM, 0, mlp_w1, DDIM, 0, 1, big, 2 * DDIM, 0, 1,
       NTOT, 2 * DDIM, DDIM, 1.f, mlp_b1, nullptr, 0, nullptr, nullptr, 1, 0);
  gemm(stream, big, nullptr, 2 * DDIM, 0, mlp_w2, 2 * DDIM, 0, 1, hloc, DDIM, 0, 1,
       NTOT, DDIM, 2 * DDIM, 1.f, mlp_b2, hloc, DDIM, bn3g, bn3b, 1, 0);

  // ============ two GAT branches ============
  float* xp[2]   = {xp1, xp2};
  float* xout[2] = {x11, x22};
  for (int br = 0; br < 2; ++br) {
    gemm(stream, hloc, nullptr, DDIM, 0, gatw[br], DDIM, 0, 1, xp[br], HDIM, 0, 1,
         NTOT, HDIM, DDIM, 1.f, nullptr, nullptr, 0, nullptr, nullptr, 0, 0);
    gat_nodescore<<<g1d(NTOT), 256, 0, stream>>>(xp[br], gatas[br], gatad[br], as_, ad_);
    fill_u<<<g1d(NTOT), 256, 0, stream>>>(mEnc, 0u, NTOT);
    fill_f<<<g1d(NTOT), 256, 0, stream>>>(denom, 0.f, NTOT);
    fill_f<<<g1d(NH), 256, 0, stream>>>(xout[br], 0.f, NH);
    gat_logit<<<g1d(NLOOP), 256, 0, stream>>>(ei, as_, ad_, exbuf, mEnc);
    gat_exp<<<g1d(NLOOP), 256, 0, stream>>>(ei, exbuf, mEnc, denom);
    gat_accum<<<g1d((long)NLOOP * 8), 256, 0, stream>>>(ei, exbuf, denom, xp[br], xout[br]);
    gat_finish<<<g1d(NH), 256, 0, stream>>>(xout[br], gatb[br]);
  }

  // ============ cross attention (8 heads, hd=8) ============
  gemm(stream, x11, nullptr, HDIM, 0, mha_iw, HDIM, 0, 1, q2, HDIM, 0, 1,
       NTOT, HDIM, HDIM, 1.f, mha_ib, nullptr, 0, nullptr, nullptr, 0, 0);
  gemm(stream, x22, nullptr, HDIM, 0, mha_iw + 64 * 64, HDIM, 0, 1, kv2, 2 * HDIM, 0, 1,
       NTOT, 2 * HDIM, HDIM, 1.f, mha_ib + 64, nullptr, 0, nullptr, nullptr, 0, 0);
  fill_f<<<g1d(ND), 256, 0, stream>>>(out + o_att, 0.f, ND);  // att accumulator
  for (int h = 0; h < 8; ++h) {
    gemm(stream, q2 + h * 8, nullptr, HDIM, (long)NPG * HDIM,
         kv2 + h * 8, 2 * HDIM, (long)NPG * 2 * HDIM, 1,
         big, NPG, sS, NGRAPH, NPG, NPG, 8, 0.3535533905932738f,
         nullptr, nullptr, 0, nullptr, nullptr, 0, 0);
    softmax264<<<NTOT, 256, 0, stream>>>(big, out + o_att, 0.125f);
    gemm(stream, big, nullptr, NPG, sS,
         kv2 + HDIM + h * 8, 2 * HDIM, (long)NPG * 2 * HDIM, 0,
         inb + h * 8, HDIM, (long)NPG * HDIM, NGRAPH, NPG, 8, NPG, 1.f,
         nullptr, nullptr, 0, nullptr, nullptr, 0, 0);
  }
  gemm(stream, inb, nullptr, HDIM, 0, mha_ow, HDIM, 0, 1, all1, HDIM, 0, 1,
       NTOT, HDIM, HDIM, 1.f, mha_ob, nullptr, 0, nullptr, nullptr, 0, 0);

  // ============ pool -> lin branches ============
  const long oP[2] = {o_p1, o_p2}, oS[2] = {o_s1, o_s2};
  for (int br = 0; br < 2; ++br) {
    addvec<<<g1d(NH), 256, 0, stream>>>(all1, xout[br], inb, NH);   // note: all1 both times
    pool_score<<<g1d(NTOT), 256, 0, stream>>>(inb, pool_w[br], as_);
    topk_graph<<<NGRAPH, 288, 0, stream>>>(as_, idxb, vals, out + oP[br], out + oS[br]);
    gather_flat<<<g1d((long)NGRAPH * KPOOL * HDIM), 256, 0, stream>>>(inb, idxb, vals, flat);
    gemm(stream, flat, nullptr, KPOOL * HDIM, 0, lin_w[br], KPOOL * HDIM, 0, 1,
         yh, HCDIM, 0, 1, NGRAPH, HCDIM, KPOOL * HDIM, 1.f,
         lin_b[br], nullptr, 0, nullptr, nullptr, 1, 0);
    if (br == 0)
      lin_final1<<<1, 128, 0, stream>>>(yh, fin_w[0], fin_b[0], out + o_y1);
    else
      lin_final2<<<1, 128, 0, stream>>>(yh, fin_w[1], fin_b[1], out + o_y2);
  }
}
